// SimplifiedNet_22771916604099
// MI455X (gfx1250) — matrix operand order repaired and checked by reading, not compiled, not run on hardware
//
#include <hip/hip_runtime.h>
#include <hip/hip_bf16.h>
#include <hip/hip_fp16.h>

// Sizes from the reference
#define T_  128
#define B_  32
#define E_  32
#define H_  64
#define L_  4
#define Z_  64
#define V_  32000
#define G4  256      // 4*H
#define IO_ 512      // 2*L*H

typedef __attribute__((ext_vector_type(16))) _Float16 v16h;
typedef __attribute__((ext_vector_type(8)))  float    v8f;
typedef unsigned int u32x4 __attribute__((ext_vector_type(4)));
typedef int          i32x4 __attribute__((ext_vector_type(4)));
typedef int          i32x8 __attribute__((ext_vector_type(8)));

// ---- Tensor Data Mover availability / arity (see CDNA5_HIP.md toolchain table) ----
#if defined(__has_builtin)
# if __has_builtin(__builtin_amdgcn_tensor_load_to_lds)
#  define USE_TDM 1
# else
#  define USE_TDM 0
# endif
#else
# define USE_TDM 0
#endif
#if __has_include(<hip/amd_detail/amd_gfx1250_TDM.h>)
# define TDM_ARGS6 1
#else
# define TDM_ARGS6 0
#endif

__device__ __forceinline__ float sigm_(float x){ return 1.0f/(1.0f+expf(-x)); }

// ---------------- WMMA tile helpers (CDNA5 16x16x32 f16 -> f32) ----------------
// A tile: 16 rows x 32 K, row-major source.
// half0 lanes hold K {0..15}, half1 lanes hold K {16..31}.
__device__ __forceinline__ v16h ld_a(const _Float16* src, int ld){
  int lane = threadIdx.x & 31;
  int half = lane >> 4, r = lane & 15;
  const _Float16* p = src + r*ld;
  v16h a;
#pragma unroll
  for (int i=0;i<16;++i){
    int k = i + 16*half;
    a[i] = p[k];
  }
  return a;
}
// B tile: 32 K x 16 cols built from row-major weight W[n][k]; pass src = W + n0*ld + k0.
// Lane holds column n; per sparse-B pattern: K = 16*half + i (contiguous 32B per lane).
__device__ __forceinline__ v16h ld_b(const _Float16* src, int ld){
  int lane = threadIdx.x & 31;
  int half = lane >> 4, n = lane & 15;
  const _Float16* p = src + n*ld + 16*half;
  v16h b;
#pragma unroll
  for (int i=0;i<16;++i) b[i] = p[i];
  return b;
}
// D tile store: VGPR g -> row g + 8*half, col = lane&15.
__device__ __forceinline__ void st_d(float* dst, int ld, v8f d){
  int lane = threadIdx.x & 31;
  int half = lane >> 4, n = lane & 15;
#pragma unroll
  for(int g=0; g<8; ++g) dst[(g + 8*half)*ld + n] = d[g];
}
__device__ __forceinline__ v8f wmma_(v16h a, v16h b, v8f c){
  return __builtin_amdgcn_wmma_f32_16x16x32_f16(false, a, false, b, (short)0, c, false, false);
}

// ---------------- TDM: 1-D global->LDS DMA (D# per ISA 08_async_tensor §8) ----------------
// bytes must be a multiple of 8; max 512KB (tile_dim0 is 16-bit in 8B units).
__device__ __forceinline__ void tdm_copy_1d(void* ldsDst, const void* gSrc, unsigned bytes){
#if USE_TDM
  unsigned n8  = bytes >> 3;                                     // 8-byte elements
  unsigned lds = (unsigned)(unsigned long long)ldsDst;           // low 32 bits = LDS offset
  unsigned long long ga = (unsigned long long)gSrc;
  u32x4 g0;
  g0[0] = 1u;                                                    // count=1 valid descriptor
  g0[1] = lds;                                                   // lds_addr
  g0[2] = (unsigned)(ga & 0xffffffffull);                        // global_addr[31:0]
  g0[3] = (unsigned)((ga >> 32) & 0x1ffffffull) | (2u << 30);    // addr[56:32] | type=2
  i32x8 g1;
  g1[0] = (int)(3u << 16);                                       // data_size=3 (8B)
  g1[1] = (int)((n8 & 0xffffu) << 16);                           // tensor_dim0[15:0] @bit48
  g1[2] = (int)(((n8 >> 16) & 0xffffu) | (1u << 16));            // tensor_dim0[31:16] | tensor_dim1=1
  g1[3] = (int)((n8 & 0xffffu) << 16);                           // tile_dim0 @bit112
  g1[4] = 1;                                                     // tile_dim1=1, tile_dim2=0
  g1[5] = (int)n8;                                               // tensor_dim0_stride[31:0]
  g1[6] = 0;                                                     // stride0 hi | stride1 lo
  g1[7] = 0;
  i32x4 z4 = {0,0,0,0};
#if TDM_ARGS6
  i32x8 z8 = {0,0,0,0,0,0,0,0};
  __builtin_amdgcn_tensor_load_to_lds(g0, g1, z4, z4, z8, 0);
#else
  __builtin_amdgcn_tensor_load_to_lds(g0, g1, z4, z4, 0);
#endif
#else
  (void)ldsDst; (void)gSrc; (void)bytes;
#endif
}
// Stage `halves` f16 elements into LDS (TDM on one wave, or cooperative fallback).
__device__ __forceinline__ void stage_h16(_Float16* dst, const _Float16* src, unsigned halves){
#if USE_TDM
  if(threadIdx.x == 0) tdm_copy_1d(dst, src, halves*2u);
#else
  for(unsigned i=threadIdx.x; i<halves; i+=blockDim.x) dst[i] = src[i];
#endif
}
__device__ __forceinline__ void stage_wait(){
#if USE_TDM
  if(threadIdx.x == 0) __builtin_amdgcn_s_wait_tensorcnt(0);
#endif
  __syncthreads();
}

// ---------------- Utility kernels ----------------
__global__ void k_f2h(const float* s, _Float16* d, int n){
  int i = blockIdx.x*blockDim.x + threadIdx.x;
  if(i < n) d[i] = (_Float16)s[i];
}
__global__ void k_bsum(const float* bih, const float* bhh, float* bs, int n){
  int i = blockIdx.x*blockDim.x + threadIdx.x;
  if(i < n) bs[i] = bih[i] + bhh[i];
}
// Embedding gather with row-0 zeroing; decoder forces token 1 at t==0.
__global__ void k_embed(const int* x, const float* emb, _Float16* out, int use_dec){
  int idx = blockIdx.x*blockDim.x + threadIdx.x;
  if(idx >= T_*B_*E_) return;
  int e  = idx % E_;
  int tb = idx / E_;
  int b  = tb % B_;
  int t  = tb / B_;
  int tok = x[t*B_ + b];
  if(use_dec && t == 0) tok = 1;
  float v = (tok == 0) ? 0.f : emb[tok*E_ + e];
  out[idx] = (_Float16)v;
}
__global__ void k_cond(const int* condition, const float* emb, float* cond){
  int idx = blockIdx.x*blockDim.x + threadIdx.x;
  if(idx >= B_*E_) return;
  int b = idx/E_, e = idx%E_;
  int tok = condition[b];
  cond[idx] = (tok == 0) ? 0.f : emb[tok*E_ + e];
}

// ---------------- G0 = embeddings @ Wih0^T  (M=4096, N=256, K=32) ----------------
__global__ void k_g0(const _Float16* A, const _Float16* W, float* C){
  int wave = threadIdx.x >> 5;
  int mt = blockIdx.y*4 + wave;   // 0..255
  int nt = blockIdx.x;            // 0..15
  v16h a = ld_a(A + mt*16*E_, E_);
  v16h b = ld_b(W + nt*16*E_, E_);
  v8f  c = {};
  c = wmma_(a, b, c);
  st_d(C + mt*16*G4 + nt*16, G4, c);
}

// ---------------- Encoder LSTM recurrence ----------------
// Single persistent workgroup. Weights TDM-staged to LDS once; h state resident as f16.
// Shared (h,c) threaded through layers; padding tokens keep old state.
__global__ void k_enc(const int* __restrict__ x, const float* __restrict__ G0,
                      const _Float16* __restrict__ WihR, const _Float16* __restrict__ Whh,
                      const float* __restrict__ bsum, float* __restrict__ hf, float* __restrict__ cf){
  extern __shared__ char smem[];
  _Float16* wWhh  = (_Float16*)smem;                   // L*256*64 halves (128KB)
  _Float16* wWihR = wWhh + L_*G4*H_;                   // 3*256*64 halves (96KB)
  float*    cS    = (float*)(wWihR + (L_-1)*G4*H_);    // 32*64 f32
  float*    gS    = cS + B_*H_;                        // 32*256 f32 gates
  _Float16* hH    = (_Float16*)(gS + B_*G4);           // 32*64 f16 (persistent h)
  _Float16* xH    = hH + B_*H_;                        // 32*64 f16 (layer input)

  int tid = threadIdx.x, wave = tid >> 5;
  stage_h16(wWhh,  Whh,  L_*G4*H_);
  stage_h16(wWihR, WihR, (L_-1)*G4*H_);
  for(int i=tid;i<B_*H_;i+=blockDim.x){ cS[i]=0.f; hH[i]=(_Float16)0.f; }
  stage_wait();

  for(int t=0;t<T_;++t){
    for(int l=0;l<L_;++l){
      const _Float16* Wh = wWhh + l*G4*H_;
      const _Float16* Wx = wWihR + (l-1)*G4*H_;
      // gates(32x256) = h @ Whh[l]^T (+ xin @ WihR[l-1]^T for l>0); 32 tiles, 4 per wave.
      for(int tt=wave*4; tt<wave*4+4; ++tt){
        int mt = tt >> 4, nt = tt & 15;
        v8f acc = {};
#pragma unroll
        for(int kk=0;kk<2;++kk){
          v16h a = ld_a(hH + mt*16*H_ + kk*32, H_);
          v16h b = ld_b(Wh + nt*16*H_ + kk*32, H_);
          acc = wmma_(a, b, acc);
        }
        if(l>0){
#pragma unroll
          for(int kk=0;kk<2;++kk){
            v16h a = ld_a(xH + mt*16*H_ + kk*32, H_);
            v16h b = ld_b(Wx + nt*16*H_ + kk*32, H_);
            acc = wmma_(a, b, acc);
          }
        }
        st_d(gS + mt*16*G4 + nt*16, G4, acc);
      }
      __syncthreads();
      for(int i=tid;i<B_*H_;i+=blockDim.x){
        int b = i / H_, j = i % H_;
        const float* gb = gS + b*G4;
        const float* bs = bsum + l*G4;
        float gi = gb[j]      + bs[j];
        float gf = gb[H_+j]   + bs[H_+j];
        float gg = gb[2*H_+j] + bs[2*H_+j];
        float go = gb[3*H_+j] + bs[3*H_+j];
        if(l==0){
          const float* g0 = G0 + (t*B_+b)*G4;
          gi += g0[j]; gf += g0[H_+j]; gg += g0[2*H_+j]; go += g0[3*H_+j];
        }
        float c2 = sigm_(gf)*cS[i] + sigm_(gi)*tanhf(gg);
        float hn = sigm_(go)*tanhf(c2);
        xH[i] = (_Float16)hn;                 // next layer's input is UNMASKED hn
        if(x[t*B_+b] != 0){ hH[i] = (_Float16)hn; cS[i] = c2; }
      }
      __syncthreads();
    }
  }
  for(int i=tid;i<B_*H_;i+=blockDim.x){ hf[i]=(float)hH[i]; cf[i]=cS[i]; }
}

// ---------------- Decoder LSTM recurrence (per-layer states, no mask) ----------------
__global__ void k_dec(const float* __restrict__ rec, const float* __restrict__ G0,
                      const _Float16* __restrict__ WihR, const _Float16* __restrict__ Whh,
                      const float* __restrict__ bsum, _Float16* __restrict__ outs_h){
  extern __shared__ char smem[];
  _Float16* wWhh  = (_Float16*)smem;                   // 128KB
  _Float16* wWihR = wWhh + L_*G4*H_;                   // 96KB
  float*    cS    = (float*)(wWihR + (L_-1)*G4*H_);    // 4*32*64 f32 (32KB)
  float*    gS    = cS + L_*B_*H_;                     // 32*256 f32 (32KB)
  _Float16* hH    = (_Float16*)(gS + B_*G4);           // 4*32*64 f16 (16KB)
  _Float16* xH    = hH + L_*B_*H_;                     // 32*64 f16

  int tid = threadIdx.x, wave = tid >> 5;
  stage_h16(wWhh,  Whh,  L_*G4*H_);
  stage_h16(wWihR, WihR, (L_-1)*G4*H_);
  for(int i=tid;i<L_*B_*H_;i+=blockDim.x){
    int l = i/(B_*H_); int r = i%(B_*H_); int b = r/H_, j = r%H_;
    hH[i] = (_Float16)rec[b*IO_ + l*H_ + j];
    cS[i] = rec[b*IO_ + L_*H_ + l*H_ + j];
  }
  stage_wait();

  for(int t=0;t<T_;++t){
    for(int l=0;l<L_;++l){
      const _Float16* Wh = wWhh + l*G4*H_;
      const _Float16* Wx = wWihR + (l-1)*G4*H_;
      _Float16* hl = hH + l*B_*H_;
      float*    cl = cS + l*B_*H_;
      for(int tt=wave*4; tt<wave*4+4; ++tt){
        int mt = tt >> 4, nt = tt & 15;
        v8f acc = {};
#pragma unroll
        for(int kk=0;kk<2;++kk){
          v16h a = ld_a(hl + mt*16*H_ + kk*32, H_);
          v16h b = ld_b(Wh + nt*16*H_ + kk*32, H_);
          acc = wmma_(a, b, acc);
        }
        if(l>0){
#pragma unroll
          for(int kk=0;kk<2;++kk){
            v16h a = ld_a(xH + mt*16*H_ + kk*32, H_);
            v16h b = ld_b(Wx + nt*16*H_ + kk*32, H_);
            acc = wmma_(a, b, acc);
          }
        }
        st_d(gS + mt*16*G4 + nt*16, G4, acc);
      }
      __syncthreads();
      for(int i=tid;i<B_*H_;i+=blockDim.x){
        int b = i / H_, j = i % H_;
        const float* gb = gS + b*G4;
        const float* bs = bsum + l*G4;
        float gi = gb[j]      + bs[j];
        float gf = gb[H_+j]   + bs[H_+j];
        float gg = gb[2*H_+j] + bs[2*H_+j];
        float go = gb[3*H_+j] + bs[3*H_+j];
        if(l==0){
          const float* g0 = G0 + (t*B_+b)*G4;
          gi += g0[j]; gf += g0[H_+j]; gg += g0[2*H_+j]; go += g0[3*H_+j];
        }
        float c2 = sigm_(gf)*cl[i] + sigm_(gi)*tanhf(gg);
        float hn = sigm_(go)*tanhf(c2);
        hl[i] = (_Float16)hn; cl[i] = c2; xH[i] = (_Float16)hn;
        if(l == L_-1) outs_h[(t*B_+b)*H_ + j] = (_Float16)hn;
      }
      __syncthreads();
    }
  }
}

// ---------------- Small generic FC: Y[m][n] = act(X[m,:K] . W[n,:K] + b[n]) ----------------
__global__ void k_fc(const float* X, const float* W, const float* bias, float* Y,
                     int M, int N, int K, int doRelu){
  int idx = blockIdx.x*blockDim.x + threadIdx.x;
  if(idx >= M*N) return;
  int m = idx/N, n = idx%N;
  const float* xp = X + m*K; const float* wp = W + n*K;
  float s = bias ? bias[n] : 0.f;
  for(int k=0;k<K;++k) s += xp[k]*wp[k];
  Y[idx] = doRelu ? fmaxf(s, 0.f) : s;
}

// First CVAE layer: input is hc_flat = [hf x4, cf x4, cond] (544) per batch row.
__global__ void k_mlp1(const float* hf, const float* cf, const float* cond,
                       const float* W1, const float* b1, float* out){
  int idx = blockIdx.x*blockDim.x + threadIdx.x;
  if(idx >= B_*544) return;
  int b = idx / 544, n = idx % 544;
  const float* w = W1 + n*544;
  float s = b1[n];
  for(int k=0;k<256;++k) s += hf[b*H_ + (k & 63)] * w[k];
  for(int k=0;k<256;++k) s += cf[b*H_ + (k & 63)] * w[256+k];
  for(int k=0;k<32;++k)  s += cond[b*E_ + k]      * w[512+k];
  out[idx] = fmaxf(s, 0.f);
}

// z = mean + eps*exp(0.5*logvar); emit the 5 small outputs; build in2=[z,cond].
__global__ void k_z(const float* mv, const float* pr, const float* eps, const float* cond,
                    float* out_tail, float* in2){
  int idx = blockIdx.x*blockDim.x + threadIdx.x;
  if(idx >= B_*Z_) return;
  int b = idx/Z_, j = idx%Z_;
  float mean = mv[b*128 + j], logv = mv[b*128 + 64 + j];
  float z = mean + eps[idx]*expf(0.5f*logv);
  out_tail[0*B_*Z_ + idx] = z;
  out_tail[1*B_*Z_ + idx] = mean;
  out_tail[2*B_*Z_ + idx] = logv;
  out_tail[3*B_*Z_ + idx] = pr[b*128 + j];
  out_tail[4*B_*Z_ + idx] = pr[b*128 + 64 + j];
  in2[b*96 + j] = z;
  if(j < E_) in2[b*96 + 64 + j] = cond[b*E_ + j];
}

// ---------------- logits = outs @ fc_W^T + fc_b  (M=4096, N=32000, K=64) ----------------
// grid(250, 64), block 128 (4 waves). TDM-stages the 16KB weight tile + 8KB activation
// tile into LDS, then each wave owns one 16-row tile and loops 8 N-subtiles.
__global__ void k_logits(const _Float16* __restrict__ A, const _Float16* __restrict__ W,
                         const float* __restrict__ bias, float* __restrict__ C){
  extern __shared__ char smem[];
  _Float16* Wl = (_Float16*)smem;       // 128 x 64 halves (16KB)
  _Float16* Al = Wl + 128*H_;           // 64 x 64 halves (8KB)
  int n0 = blockIdx.x*128;
  int m0 = blockIdx.y*64;
  stage_h16(Wl, W + (size_t)n0*H_, 128*H_);
  stage_h16(Al, A + (size_t)m0*H_, 64*H_);
  stage_wait();

  int wave = threadIdx.x >> 5;
  int lane = threadIdx.x & 31;
  const _Float16* Arow = Al + wave*16*H_;
  v16h a0 = ld_a(Arow,      H_);
  v16h a1 = ld_a(Arow + 32, H_);
  float* Crow = C + (size_t)(m0 + wave*16)*V_ + n0;
#pragma unroll
  for(int nt=0; nt<8; ++nt){
    int n = nt*16;
    v8f acc = {};
    acc = wmma_(a0, ld_b(Wl + n*H_,      H_), acc);
    acc = wmma_(a1, ld_b(Wl + n*H_ + 32, H_), acc);
    float bv = bias[n0 + n + (lane & 15)];
#pragma unroll
    for(int g=0; g<8; ++g) acc[g] += bv;
    st_d(Crow + n, V_, acc);
  }
}

// ---------------- Host orchestration ----------------
extern "C" void kernel_launch(void* const* d_in, const int* in_sizes, int n_in,
                              void* d_out, int out_size, void* d_ws, size_t ws_size,
                              hipStream_t stream){
  const int*   x       = (const int*)  d_in[0];
  const int*   cndtok  = (const int*)  d_in[1];
  const float* eps     = (const float*)d_in[2];
  const float* emb_enc = (const float*)d_in[3];
  const float* eWih0   = (const float*)d_in[4];
  const float* eWihR   = (const float*)d_in[5];
  const float* eWhh    = (const float*)d_in[6];
  const float* ebih    = (const float*)d_in[7];
  const float* ebhh    = (const float*)d_in[8];
  const float* cvemb   = (const float*)d_in[9];
  const float* ceW1    = (const float*)d_in[10];
  const float* ceb1    = (const float*)d_in[11];
  const float* ceW2    = (const float*)d_in[12];
  const float* ceb2    = (const float*)d_in[13];
  const float* cpW1    = (const float*)d_in[14];
  const float* cpb1    = (const float*)d_in[15];
  const float* cpW2    = (const float*)d_in[16];
  const float* cpb2    = (const float*)d_in[17];
  const float* cdW1    = (const float*)d_in[18];
  const float* cdb1    = (const float*)d_in[19];
  const float* cdW2    = (const float*)d_in[20];
  const float* cdb2    = (const float*)d_in[21];
  const float* emb_dec = (const float*)d_in[22];
  const float* dWih0   = (const float*)d_in[23];
  const float* dWihR   = (const float*)d_in[24];
  const float* dWhh    = (const float*)d_in[25];
  const float* dbih    = (const float*)d_in[26];
  const float* dbhh    = (const float*)d_in[27];
  const float* fcW     = (const float*)d_in[28];
  const float* fcb     = (const float*)d_in[29];
  float* dout = (float*)d_out;

  // Workspace bump allocator (~14 MB total)
  char* wptr = (char*)d_ws;
  auto alloc = [&](size_t bytes)->char*{
    char* p = wptr; wptr += (bytes + 255) & ~size_t(255); return p;
  };
  _Float16* exh     = (_Float16*)alloc((size_t)T_*B_*E_*2);
  _Float16* dexh    = (_Float16*)alloc((size_t)T_*B_*E_*2);
  _Float16* eWih0h  = (_Float16*)alloc((size_t)G4*E_*2);
  _Float16* eWihRh  = (_Float16*)alloc((size_t)(L_-1)*G4*H_*2);
  _Float16* eWhhh   = (_Float16*)alloc((size_t)L_*G4*H_*2);
  _Float16* dWih0h  = (_Float16*)alloc((size_t)G4*E_*2);
  _Float16* dWihRh  = (_Float16*)alloc((size_t)(L_-1)*G4*H_*2);
  _Float16* dWhhh   = (_Float16*)alloc((size_t)L_*G4*H_*2);
  _Float16* fcWh    = (_Float16*)alloc((size_t)V_*H_*2);
  float* bsumE  = (float*)alloc((size_t)L_*G4*4);
  float* bsumD  = (float*)alloc((size_t)L_*G4*4);
  float* G0enc  = (float*)alloc((size_t)T_*B_*G4*4);
  float* G0dec  = (float*)alloc((size_t)T_*B_*G4*4);
  float* hf     = (float*)alloc((size_t)B_*H_*4);
  float* cf     = (float*)alloc((size_t)B_*H_*4);
  float* condb  = (float*)alloc((size_t)B_*E_*4);
  float* relu1  = (float*)alloc((size_t)B_*544*4);
  float* ph     = (float*)alloc((size_t)B_*128*4);
  float* mvb    = (float*)alloc((size_t)B_*128*4);
  float* prb    = (float*)alloc((size_t)B_*128*4);
  float* in2    = (float*)alloc((size_t)B_*96*4);
  float* t2     = (float*)alloc((size_t)B_*IO_*4);
  float* recb   = (float*)alloc((size_t)B_*IO_*4);
  _Float16* outsh = (_Float16*)alloc((size_t)T_*B_*H_*2);

  // LDS budgets (dynamic; CDNA5 allows up to 320KB per workgroup)
  size_t wBytes   = (size_t)(L_ + (L_-1))*G4*H_*2;                         // 229376
  size_t encSmem  = wBytes + (size_t)(B_*H_ + B_*G4)*4 + (size_t)2*B_*H_*2;          // ~272KB
  size_t decSmem  = wBytes + (size_t)(L_*B_*H_ + B_*G4)*4
                  + (size_t)(L_+1)*B_*H_*2;                                           // ~308KB
  size_t logSmem  = (size_t)(128 + 64)*H_*2;                                          // 24KB
  (void)hipFuncSetAttribute((const void*)k_enc,    hipFuncAttributeMaxDynamicSharedMemorySize, (int)encSmem);
  (void)hipFuncSetAttribute((const void*)k_dec,    hipFuncAttributeMaxDynamicSharedMemorySize, (int)decSmem);
  (void)hipFuncSetAttribute((const void*)k_logits, hipFuncAttributeMaxDynamicSharedMemorySize, (int)logSmem);

  auto cvt = [&](const float* s, _Float16* d, int n){
    k_f2h<<<dim3((n+255)/256), dim3(256), 0, stream>>>(s, d, n);
  };
  // 1) Weight conversion f32 -> f16
  cvt(eWih0, eWih0h, G4*E_);
  cvt(eWihR, eWihRh, (L_-1)*G4*H_);
  cvt(eWhh,  eWhhh,  L_*G4*H_);
  cvt(dWih0, dWih0h, G4*E_);
  cvt(dWihR, dWihRh, (L_-1)*G4*H_);
  cvt(dWhh,  dWhhh,  L_*G4*H_);
  cvt(fcW,   fcWh,   V_*H_);
  k_bsum<<<dim3(4), dim3(256), 0, stream>>>(ebih, ebhh, bsumE, L_*G4);
  k_bsum<<<dim3(4), dim3(256), 0, stream>>>(dbih, dbhh, bsumD, L_*G4);

  // 2) Embedding gathers (f16)
  k_embed<<<dim3((T_*B_*E_+255)/256), dim3(256), 0, stream>>>(x, emb_enc, exh,  0);
  k_embed<<<dim3((T_*B_*E_+255)/256), dim3(256), 0, stream>>>(x, emb_dec, dexh, 1);

  // 3) Time-parallel input GEMMs: G0 = emb @ Wih0^T  (WMMA)
  k_g0<<<dim3(16, 64), dim3(128), 0, stream>>>(exh,  eWih0h, G0enc);
  k_g0<<<dim3(16, 64), dim3(128), 0, stream>>>(dexh, dWih0h, G0dec);

  // 4) Encoder recurrence (persistent workgroup, TDM-staged weights, WMMA per step)
  k_enc<<<dim3(1), dim3(256), encSmem, stream>>>(x, G0enc, eWihRh, eWhhh, bsumE, hf, cf);

  // 5) CVAE head (tiny MLPs)
  k_cond<<<dim3(4), dim3(256), 0, stream>>>(cndtok, cvemb, condb);
  k_mlp1<<<dim3((B_*544+255)/256), dim3(256), 0, stream>>>(hf, cf, condb, ceW1, ceb1, relu1);
  k_fc<<<dim3((B_*128+255)/256), dim3(256), 0, stream>>>(relu1, ceW2, ceb2, mvb, B_, 128, 544, 0);
  k_fc<<<dim3((B_*128+255)/256), dim3(256), 0, stream>>>(condb, cpW1, cpb1, ph,  B_, 128, 32,  1);
  k_fc<<<dim3((B_*128+255)/256), dim3(256), 0, stream>>>(ph,    cpW2, cpb2, prb, B_, 128, 128, 0);
  k_z<<<dim3((B_*Z_+255)/256), dim3(256), 0, stream>>>(mvb, prb, eps, condb,
                                                       dout + (size_t)T_*B_*V_, in2);
  k_fc<<<dim3((B_*IO_+255)/256), dim3(256), 0, stream>>>(in2, cdW1, cdb1, t2,   B_, IO_, 96,  1);
  k_fc<<<dim3((B_*IO_+255)/256), dim3(256), 0, stream>>>(t2,  cdW2, cdb2, recb, B_, IO_, IO_, 0);

  // 6) Decoder recurrence
  k_dec<<<dim3(1), dim3(256), decSmem, stream>>>(recb, G0dec, dWihRh, dWhhh, bsumD, outsh);

  // 7) Output projection (write-BW bound; f16 WMMA, f32 accumulate, TDM-staged tiles)
  k_logits<<<dim3(250, 64), dim3(128), logSmem, stream>>>(outsh, fcWh, fcb, dout);
}
